// Transformer_22368189677832
// MI455X (gfx1250) — compile-verified, matrix operand order checked by reading, not run on hardware
//
#include <hip/hip_runtime.h>
#include <hip/hip_bf16.h>
#include <math.h>

typedef __attribute__((ext_vector_type(16))) _Float16 v16h;
typedef __attribute__((ext_vector_type(8)))  _Float16 v8h;
typedef __attribute__((ext_vector_type(4)))  _Float16 v4h;
typedef __attribute__((ext_vector_type(2)))  _Float16 v2h;
typedef __attribute__((ext_vector_type(8)))  float    v8f;

#define NB_B   4
#define NB_S   512
#define NB_D   1024
#define NB_F   4096
#define NB_H   16
#define NB_DK  64
#define NB_L   4
#define LN_EPS 1e-5f

// ---------------------------------------------------------------------------
// WMMA A/B fragment loader: 16x32 f16 fragment for lane (half = lane>>4,
// l16 = lane&15).  Lane holds row l16, K values {half*8..+7} U {16+half*8..+7}
// -> two 16B LDS reads. Row pitch 40 f16 (80B): 16B-aligned, conflict-free.
// ---------------------------------------------------------------------------
__device__ __forceinline__ v16h ldfrag(const _Float16* row, int half) {
    v8h lo = *(const v8h*)(row + half * 8);
    v8h hi = *(const v8h*)(row + 16 + half * 8);
    v16h r;
#pragma unroll
    for (int i = 0; i < 8; ++i) { r[i] = lo[i]; r[i + 8] = hi[i]; }
    return r;
}

// ---------------------------------------------------------------------------
// Split-phase tile staging (f32 global -> f16 LDS).
// Phase 1 issues all global float4 loads into DISTINCT registers (no waits
// back-to-back); phase 2 (placed after the WMMA block) converts and stores
// to LDS, so global latency is hidden behind the matrix ops.
// ---------------------------------------------------------------------------

// 128 rows x 32 k, k-contiguous.  base = &SRC[row0*ld + k0].
struct StA { float4 f[4]; };
__device__ __forceinline__ void loadA(StA& r, const float* base, size_t ld,
                                      int tid) {
#pragma unroll
    for (int it = 0; it < 4; ++it) {
        int idx = tid + it * 256;
        int m = idx >> 3, c4 = idx & 7;
        r.f[it] = *(const float4*)(base + (size_t)m * ld + c4 * 4);
    }
}
__device__ __forceinline__ void storeA(const StA& r, _Float16 (*dst)[40],
                                       int tid) {
#pragma unroll
    for (int it = 0; it < 4; ++it) {
        int idx = tid + it * 256;
        int m = idx >> 3, c4 = idx & 7;
        v4h h;
        h[0] = (_Float16)r.f[it].x; h[1] = (_Float16)r.f[it].y;
        h[2] = (_Float16)r.f[it].z; h[3] = (_Float16)r.f[it].w;
        *(v4h*)&dst[m][c4 * 4] = h;
    }
}

// Transposed: 32 k-rows x 128 n-cols of W (n-contiguous) -> dst[n][k].
struct StW { float4 f[2][2]; };
__device__ __forceinline__ void loadW(StW& r, const float* base, size_t ld,
                                      int tid) {
#pragma unroll
    for (int it = 0; it < 2; ++it) {
        int idx = tid + it * 256;
        int n4 = idx & 31, k2 = idx >> 5;
        const float* p0 = base + (size_t)(2 * k2) * ld + n4 * 4;
        r.f[it][0] = *(const float4*)p0;
        r.f[it][1] = *(const float4*)(p0 + ld);
    }
}
__device__ __forceinline__ void storeW(const StW& r, _Float16 (*dst)[40],
                                       int tid) {
#pragma unroll
    for (int it = 0; it < 2; ++it) {
        int idx = tid + it * 256;
        int n4 = idx & 31, k2 = idx >> 5;
        v2h q;
        q[0] = (_Float16)r.f[it][0].x; q[1] = (_Float16)r.f[it][1].x;
        *(v2h*)&dst[n4 * 4 + 0][2 * k2] = q;
        q[0] = (_Float16)r.f[it][0].y; q[1] = (_Float16)r.f[it][1].y;
        *(v2h*)&dst[n4 * 4 + 1][2 * k2] = q;
        q[0] = (_Float16)r.f[it][0].z; q[1] = (_Float16)r.f[it][1].z;
        *(v2h*)&dst[n4 * 4 + 2][2 * k2] = q;
        q[0] = (_Float16)r.f[it][0].w; q[1] = (_Float16)r.f[it][1].w;
        *(v2h*)&dst[n4 * 4 + 3][2 * k2] = q;
    }
}

// Transposed: 32 k-rows x 64 n-cols of V -> dst[n][k]. 256 threads = 1 pass.
struct StV { float4 f[2]; };
__device__ __forceinline__ void loadV(StV& r, const float* base, size_t ld,
                                      int tid) {
    int n4 = tid & 15, k2 = tid >> 4;
    const float* p0 = base + (size_t)(2 * k2) * ld + n4 * 4;
    r.f[0] = *(const float4*)p0;
    r.f[1] = *(const float4*)(p0 + ld);
}
__device__ __forceinline__ void storeV(const StV& r, _Float16 (*dst)[40],
                                       int tid) {
    int n4 = tid & 15, k2 = tid >> 4;
    v2h q;
    q[0] = (_Float16)r.f[0].x; q[1] = (_Float16)r.f[1].x;
    *(v2h*)&dst[n4 * 4 + 0][2 * k2] = q;
    q[0] = (_Float16)r.f[0].y; q[1] = (_Float16)r.f[1].y;
    *(v2h*)&dst[n4 * 4 + 1][2 * k2] = q;
    q[0] = (_Float16)r.f[0].z; q[1] = (_Float16)r.f[1].z;
    *(v2h*)&dst[n4 * 4 + 2][2 * k2] = q;
    q[0] = (_Float16)r.f[0].w; q[1] = (_Float16)r.f[1].w;
    *(v2h*)&dst[n4 * 4 + 3][2 * k2] = q;
}

// ---------------------------------------------------------------------------
// Generic GEMM:  C[M,N] = act( A[M,K] * W[K,N] + bias[N] )
// 128x128 block tile, K-step 32, 256 threads (8 wave32), wave tile 32x64.
// LDS double-buffered; next-tile global loads issued before the WMMAs,
// converts/LDS-stores after them.
// ---------------------------------------------------------------------------
__global__ __launch_bounds__(256)
void gemm_kernel(const float* __restrict__ A, const float* __restrict__ W,
                 const float* __restrict__ bias, float* __restrict__ C,
                 int M, int Kd, int N, int relu) {
    __shared__ _Float16 sA[2][128][40];
    __shared__ _Float16 sB[2][128][40];   // sB[n][k]
    const int tid  = threadIdx.x;
    const int n0   = blockIdx.x * 128;
    const int m0   = blockIdx.y * 128;
    const int lane = tid & 31, wave = tid >> 5;
    const int wm   = (wave & 3) * 32;
    const int wn   = (wave >> 2) * 64;
    const int half = lane >> 4, l16 = lane & 15;

    const float* Abase = A + (size_t)m0 * Kd;
    const float* Wbase = W + n0;

    v8f z = {};
    v8f acc[2][4];
#pragma unroll
    for (int mi = 0; mi < 2; ++mi)
#pragma unroll
        for (int ni = 0; ni < 4; ++ni) acc[mi][ni] = z;

    const int nk = Kd >> 5;
    StA ra; StW rw;
    loadA(ra, Abase, Kd, tid);
    loadW(rw, Wbase, N,  tid);
    storeA(ra, sA[0], tid);
    storeW(rw, sB[0], tid);
    int cur = 0;
    for (int kt = 0; kt < nk; ++kt) {
        __syncthreads();
        const bool pf = (kt + 1 < nk);
        if (pf) {                                  // issue next-tile loads
            int k0n = (kt + 1) << 5;
            loadA(ra, Abase + k0n, Kd, tid);
            loadW(rw, Wbase + (size_t)k0n * N, N, tid);
        }
        v16h af[2], bf[4];
#pragma unroll
        for (int mi = 0; mi < 2; ++mi)
            af[mi] = ldfrag(&sA[cur][wm + mi * 16 + l16][0], half);
#pragma unroll
        for (int ni = 0; ni < 4; ++ni)
            bf[ni] = ldfrag(&sB[cur][wn + ni * 16 + l16][0], half);
#pragma unroll
        for (int mi = 0; mi < 2; ++mi)
#pragma unroll
            for (int ni = 0; ni < 4; ++ni)
                acc[mi][ni] = __builtin_amdgcn_wmma_f32_16x16x32_f16(
                    false, af[mi], false, bf[ni], (short)0, acc[mi][ni], false, false);
        if (pf) {                                  // latency now hidden
            storeA(ra, sA[cur ^ 1], tid);
            storeW(rw, sB[cur ^ 1], tid);
        }
        cur ^= 1;
    }

#pragma unroll
    for (int mi = 0; mi < 2; ++mi)
#pragma unroll
        for (int ni = 0; ni < 4; ++ni) {
            int col = n0 + wn + ni * 16 + l16;
            float bv = bias[col];
#pragma unroll
            for (int r = 0; r < 8; ++r) {
                int row = m0 + wm + mi * 16 + half * 8 + r;
                float v = acc[mi][ni][r] + bv;
                if (relu) v = fmaxf(v, 0.0f);
                C[(size_t)row * N + col] = v;
            }
        }
}

// ---------------------------------------------------------------------------
// scores[bh,q,k] = (Q_bh[q,:] . K_bh[k,:]) * scale.  K rows are B^T already,
// so both operands stage k-contiguously.   grid: (Sk/128, Sq/128, B*H)
// ---------------------------------------------------------------------------
__global__ __launch_bounds__(256)
void attn_qk_kernel(const float* __restrict__ Q, const float* __restrict__ K,
                    float* __restrict__ SC, float scale) {
    __shared__ _Float16 sA[2][128][40];
    __shared__ _Float16 sB[2][128][40];
    const int tid = threadIdx.x;
    const int bh  = blockIdx.z;
    const int b   = bh >> 4, h = bh & 15;
    const int n0 = blockIdx.x * 128, m0 = blockIdx.y * 128;
    const float* qb = Q + (size_t)b * NB_S * NB_D + h * NB_DK + (size_t)m0 * NB_D;
    const float* kb = K + (size_t)b * NB_S * NB_D + h * NB_DK + (size_t)n0 * NB_D;
    float* out = SC + (size_t)bh * NB_S * NB_S;
    const int lane = tid & 31, wave = tid >> 5;
    const int wm = (wave & 3) * 32, wn = (wave >> 2) * 64;
    const int half = lane >> 4, l16 = lane & 15;

    v8f z = {};
    v8f acc[2][4];
#pragma unroll
    for (int mi = 0; mi < 2; ++mi)
#pragma unroll
        for (int ni = 0; ni < 4; ++ni) acc[mi][ni] = z;

    StA ra, rb;
    loadA(ra, qb, NB_D, tid);
    loadA(rb, kb, NB_D, tid);
    storeA(ra, sA[0], tid);
    storeA(rb, sB[0], tid);
    int cur = 0;
#pragma unroll
    for (int kt = 0; kt < NB_DK / 32; ++kt) {
        __syncthreads();
        const bool pf = (kt + 1 < NB_DK / 32);
        if (pf) {
            int k0n = (kt + 1) << 5;
            loadA(ra, qb + k0n, NB_D, tid);
            loadA(rb, kb + k0n, NB_D, tid);
        }
        v16h af[2], bf[4];
#pragma unroll
        for (int mi = 0; mi < 2; ++mi)
            af[mi] = ldfrag(&sA[cur][wm + mi * 16 + l16][0], half);
#pragma unroll
        for (int ni = 0; ni < 4; ++ni)
            bf[ni] = ldfrag(&sB[cur][wn + ni * 16 + l16][0], half);
#pragma unroll
        for (int mi = 0; mi < 2; ++mi)
#pragma unroll
            for (int ni = 0; ni < 4; ++ni)
                acc[mi][ni] = __builtin_amdgcn_wmma_f32_16x16x32_f16(
                    false, af[mi], false, bf[ni], (short)0, acc[mi][ni], false, false);
        if (pf) {
            storeA(ra, sA[cur ^ 1], tid);
            storeA(rb, sB[cur ^ 1], tid);
        }
        cur ^= 1;
    }

#pragma unroll
    for (int mi = 0; mi < 2; ++mi)
#pragma unroll
        for (int ni = 0; ni < 4; ++ni) {
            int col = n0 + wn + ni * 16 + l16;
#pragma unroll
            for (int r = 0; r < 8; ++r) {
                int row = m0 + wm + mi * 16 + half * 8 + r;
                out[(size_t)row * NB_S + col] = acc[mi][ni][r] * scale;
            }
        }
}

// ---------------------------------------------------------------------------
// masked softmax over keys (512), in place. Reproduces where(mask,s,-1e9).
// ---------------------------------------------------------------------------
__global__ __launch_bounds__(256)
void softmax_kernel(float* __restrict__ SC, const int* __restrict__ keytok,
                    int causal) {
    const int row = blockIdx.x;          // B*H*S rows
    const int q   = row & (NB_S - 1);
    const int bh  = row >> 9;
    const int b   = bh >> 4;
    float* p = SC + (size_t)row * NB_S;
    const int tid = threadIdx.x;
    __shared__ float red[256];

    int j0 = tid, j1 = tid + 256;
    bool v0 = true, v1 = true;
    if (keytok) { v0 = keytok[b * NB_S + j0] != 0; v1 = keytok[b * NB_S + j1] != 0; }
    if (causal) { v0 = v0 && (j0 <= q); v1 = v1 && (j1 <= q); }
    float s0 = v0 ? p[j0] : -1e9f;
    float s1 = v1 ? p[j1] : -1e9f;

    red[tid] = fmaxf(s0, s1); __syncthreads();
    for (int s = 128; s > 0; s >>= 1) {
        if (tid < s) red[tid] = fmaxf(red[tid], red[tid + s]);
        __syncthreads();
    }
    float mx = red[0]; __syncthreads();
    float e0 = __expf(s0 - mx), e1 = __expf(s1 - mx);
    red[tid] = e0 + e1; __syncthreads();
    for (int s = 128; s > 0; s >>= 1) {
        if (tid < s) red[tid] += red[tid + s];
        __syncthreads();
    }
    float inv = 1.0f / red[0];
    p[j0] = e0 * inv; p[j1] = e1 * inv;
}

// ---------------------------------------------------------------------------
// ctx_bh[q,n] = sum_k P_bh[q,k] * V_bh[k,n]  (M=512, N=64, K=512)
// output into [B*S, D] at column h*dk.   grid: (Sq/128, B*H)
// ---------------------------------------------------------------------------
__global__ __launch_bounds__(256)
void attn_av_kernel(const float* __restrict__ P, const float* __restrict__ V,
                    float* __restrict__ OUT) {
    __shared__ _Float16 sA[2][128][40];
    __shared__ _Float16 sB[2][64][40];
    const int tid = threadIdx.x;
    const int bh  = blockIdx.y;
    const int b   = bh >> 4, h = bh & 15;
    const int m0  = blockIdx.x * 128;
    const float* pb = P + (size_t)bh * NB_S * NB_S + (size_t)m0 * NB_S;
    const float* vb = V + (size_t)b * NB_S * NB_D + h * NB_DK;
    float* ob = OUT + (size_t)b * NB_S * NB_D + h * NB_DK;
    const int lane = tid & 31, wave = tid >> 5;
    const int wm = (wave & 3) * 32, wn = (wave >> 2) * 32;
    const int half = lane >> 4, l16 = lane & 15;

    v8f z = {};
    v8f acc[2][2];
#pragma unroll
    for (int mi = 0; mi < 2; ++mi)
#pragma unroll
        for (int ni = 0; ni < 2; ++ni) acc[mi][ni] = z;

    StA ra; StV rv;
    loadA(ra, pb, NB_S, tid);
    loadV(rv, vb, NB_D, tid);
    storeA(ra, sA[0], tid);
    storeV(rv, sB[0], tid);
    int cur = 0;
    for (int kt = 0; kt < NB_S / 32; ++kt) {
        __syncthreads();
        const bool pf = (kt + 1 < NB_S / 32);
        if (pf) {
            int k0n = (kt + 1) << 5;
            loadA(ra, pb + k0n, NB_S, tid);
            loadV(rv, vb + (size_t)k0n * NB_D, NB_D, tid);
        }
        v16h af[2], bf[2];
#pragma unroll
        for (int mi = 0; mi < 2; ++mi)
            af[mi] = ldfrag(&sA[cur][wm + mi * 16 + l16][0], half);
#pragma unroll
        for (int ni = 0; ni < 2; ++ni)
            bf[ni] = ldfrag(&sB[cur][wn + ni * 16 + l16][0], half);
#pragma unroll
        for (int mi = 0; mi < 2; ++mi)
#pragma unroll
            for (int ni = 0; ni < 2; ++ni)
                acc[mi][ni] = __builtin_amdgcn_wmma_f32_16x16x32_f16(
                    false, af[mi], false, bf[ni], (short)0, acc[mi][ni], false, false);
        if (pf) {
            storeA(ra, sA[cur ^ 1], tid);
            storeV(rv, sB[cur ^ 1], tid);
        }
        cur ^= 1;
    }

#pragma unroll
    for (int mi = 0; mi < 2; ++mi)
#pragma unroll
        for (int ni = 0; ni < 2; ++ni) {
            int col = wn + ni * 16 + l16;
#pragma unroll
            for (int r = 0; r < 8; ++r) {
                int row = m0 + wm + mi * 16 + half * 8 + r;
                ob[(size_t)row * NB_D + col] = acc[mi][ni][r];
            }
        }
}

// ---------------------------------------------------------------------------
// O[row,:] = LayerNorm( alpha*R[row,:] + T[row,:] ) * g + beta   (float4 I/O)
// ---------------------------------------------------------------------------
__global__ __launch_bounds__(256)
void add_ln_kernel(const float* __restrict__ R, const float* __restrict__ T,
                   const float* __restrict__ g, const float* __restrict__ bt,
                   float* __restrict__ O, float alpha) {
    const int row = blockIdx.x;
    const int tid = threadIdx.x;
    const float4 rv = ((const float4*)(R + (size_t)row * NB_D))[tid];
    const float4 tv = ((const float4*)(T + (size_t)row * NB_D))[tid];
    float u0 = alpha * rv.x + tv.x, u1 = alpha * rv.y + tv.y;
    float u2 = alpha * rv.z + tv.z, u3 = alpha * rv.w + tv.w;
    float s1 = u0 + u1 + u2 + u3;
    float s2 = u0 * u0 + u1 * u1 + u2 * u2 + u3 * u3;

    __shared__ float ra[256], rb[256];
    ra[tid] = s1; rb[tid] = s2; __syncthreads();
    for (int s = 128; s > 0; s >>= 1) {
        if (tid < s) { ra[tid] += ra[tid + s]; rb[tid] += rb[tid + s]; }
        __syncthreads();
    }
    float mu  = ra[0] * (1.0f / NB_D);
    float var = rb[0] * (1.0f / NB_D) - mu * mu;
    float is  = rsqrtf(var + LN_EPS);

    const float4 gv = ((const float4*)g)[tid];
    const float4 bv = ((const float4*)bt)[tid];
    float4 ov;
    ov.x = (u0 - mu) * is * gv.x + bv.x;
    ov.y = (u1 - mu) * is * gv.y + bv.y;
    ov.z = (u2 - mu) * is * gv.z + bv.z;
    ov.w = (u3 - mu) * is * gv.w + bv.w;
    ((float4*)(O + (size_t)row * NB_D))[tid] = ov;
}

// ---------------------------------------------------------------------------
// O[row,:] = emb[tok[row],:] + positional_encoding(row % S, :)   (float4 I/O)
// ---------------------------------------------------------------------------
__global__ __launch_bounds__(256)
void embed_kernel(const int* __restrict__ tok, const float* __restrict__ emb,
                  float* __restrict__ O) {
    const int row = blockIdx.x;
    const int s   = row & (NB_S - 1);
    const int t   = tok[row];
    const int tid = threadIdx.x;
    const float4 ev = ((const float4*)(emb + (size_t)t * NB_D))[tid];
    const float c = -logf(10000.0f) / (float)NB_D;
    int d0 = tid * 4;
    float a0 = (float)s * __expf(c * (float)d0);
    float a1 = (float)s * __expf(c * (float)(d0 + 2));
    float4 o4;
    o4.x = ev.x + __sinf(a0);
    o4.y = ev.y + __cosf(a0);
    o4.z = ev.z + __sinf(a1);
    o4.w = ev.w + __cosf(a1);
    ((float4*)(O + (size_t)row * NB_D))[tid] = o4;
}

// ---------------------------------------------------------------------------
// host orchestration
// ---------------------------------------------------------------------------
static inline void gemm(const float* A, const float* W, const float* bias,
                        float* C, int M, int Kd, int N, int relu,
                        hipStream_t st) {
    dim3 g(N / 128, M / 128);
    gemm_kernel<<<g, 256, 0, st>>>(A, W, bias, C, M, Kd, N, relu);
}

static inline void run_mha(const float* xq, const float* xkv, const float* w,
                           const float* bvec, const int* keytok, int causal,
                           float* Qb, float* Kb, float* Vb, float* T1, float* T2,
                           float* SC, hipStream_t st) {
    const int D = NB_D, M = NB_B * NB_S;
    const size_t DD = (size_t)D * D;
    gemm(xq,  w + 0 * DD, bvec + 0 * D, Qb, M, D, D, 0, st);
    gemm(xkv, w + 1 * DD, bvec + 1 * D, Kb, M, D, D, 0, st);
    gemm(xkv, w + 2 * DD, bvec + 2 * D, Vb, M, D, D, 0, st);
    attn_qk_kernel<<<dim3(NB_S / 128, NB_S / 128, NB_B * NB_H), 256, 0, st>>>(
        Qb, Kb, SC, 0.125f);  // 1/sqrt(64)
    softmax_kernel<<<dim3(NB_B * NB_H * NB_S), 256, 0, st>>>(SC, keytok, causal);
    attn_av_kernel<<<dim3(NB_S / 128, NB_B * NB_H), 256, 0, st>>>(SC, Vb, T1);
    gemm(T1, w + 3 * DD, bvec + 3 * D, T2, M, D, D, 0, st);
}

extern "C" void kernel_launch(void* const* d_in, const int* in_sizes, int n_in,
                              void* d_out, int out_size, void* d_ws,
                              size_t ws_size, hipStream_t stream) {
    (void)in_sizes; (void)n_in; (void)out_size; (void)ws_size;
    const int*   src        = (const int*)d_in[0];
    const int*   tgt        = (const int*)d_in[1];
    const float* src_emb    = (const float*)d_in[2];
    const float* tgt_emb    = (const float*)d_in[3];
    const float* enc_attn_w = (const float*)d_in[4];
    const float* enc_attn_b = (const float*)d_in[5];
    const float* enc_w1     = (const float*)d_in[6];
    const float* enc_b1     = (const float*)d_in[7];
    const float* enc_w2     = (const float*)d_in[8];
    const float* enc_b2     = (const float*)d_in[9];
    const float* enc_ln_s   = (const float*)d_in[10];
    const float* enc_ln_b   = (const float*)d_in[11];
    const float* dec_attn_w = (const float*)d_in[12];
    const float* dec_attn_b = (const float*)d_in[13];
    const float* dec_w1     = (const float*)d_in[14];
    const float* dec_b1     = (const float*)d_in[15];
    const float* dec_w2     = (const float*)d_in[16];
    const float* dec_b2     = (const float*)d_in[17];
    const float* dec_ln_s   = (const float*)d_in[18];
    const float* dec_ln_b   = (const float*)d_in[19];

    const int M = NB_B * NB_S;              // 2048 rows
    const size_t actN = (size_t)M * NB_D;
    float* ws = (float*)d_ws;
    float* X  = ws;                         // encoder activations
    float* Y  = X + actN;                   // decoder activations
    float* Qb = Y + actN;
    float* Kb = Qb + actN;
    float* Vb = Kb + actN;
    float* T1 = Vb + actN;                  // attn context / ffn out
    float* T2 = T1 + actN;                  // attn out-projection
    float* Hf = T2 + actN;                  // ffn hidden  [M, F]
    float* SC = Hf + (size_t)M * NB_F;      // scores      [B*H, S, S]

    embed_kernel<<<dim3(M), 256, 0, stream>>>(src, src_emb, X);
    embed_kernel<<<dim3(M), 256, 0, stream>>>(tgt, tgt_emb, Y);

    // -------- encoder --------
    for (int i = 0; i < NB_L; ++i) {
        const float* w  = enc_attn_w + (size_t)i * 4 * NB_D * NB_D;
        const float* bv = enc_attn_b + (size_t)i * 4 * NB_D;
        const float* ls = enc_ln_s + (size_t)i * 2 * NB_D;
        const float* lb = enc_ln_b + (size_t)i * 2 * NB_D;
        run_mha(X, X, w, bv, src, 0, Qb, Kb, Vb, T1, T2, SC, stream);
        add_ln_kernel<<<dim3(M), 256, 0, stream>>>(X, T2, ls, lb, X, 1.0f);
        gemm(X,  enc_w1 + (size_t)i * NB_D * NB_F, enc_b1 + (size_t)i * NB_F,
             Hf, M, NB_D, NB_F, 1, stream);
        gemm(Hf, enc_w2 + (size_t)i * NB_F * NB_D, enc_b2 + (size_t)i * NB_D,
             T1, M, NB_F, NB_D, 0, stream);
        // f = ffn(x) + x, then LN(x + f)  =>  LN(2x + ffn_core)
        add_ln_kernel<<<dim3(M), 256, 0, stream>>>(X, T1, ls + NB_D, lb + NB_D,
                                                   X, 2.0f);
    }

    // -------- decoder --------
    for (int i = 0; i < NB_L; ++i) {
        const float* w0  = dec_attn_w + (size_t)i * 2 * 4 * NB_D * NB_D;
        const float* b0  = dec_attn_b + (size_t)i * 2 * 4 * NB_D;
        const float* w1c = w0 + (size_t)4 * NB_D * NB_D;
        const float* b1c = b0 + 4 * NB_D;
        const float* ls  = dec_ln_s + (size_t)i * 3 * NB_D;
        const float* lb  = dec_ln_b + (size_t)i * 3 * NB_D;
        // masked self-attention (causal + tgt pad)
        run_mha(Y, Y, w0, b0, tgt, 1, Qb, Kb, Vb, T1, T2, SC, stream);
        add_ln_kernel<<<dim3(M), 256, 0, stream>>>(Y, T2, ls, lb, Y, 1.0f);
        // cross-attention (keys/values from encoder output X, src pad mask)
        run_mha(Y, X, w1c, b1c, src, 0, Qb, Kb, Vb, T1, T2, SC, stream);
        add_ln_kernel<<<dim3(M), 256, 0, stream>>>(Y, T2, ls + NB_D, lb + NB_D,
                                                   Y, 1.0f);
        // FFN (single residual in decoder)
        gemm(Y,  dec_w1 + (size_t)i * NB_D * NB_F, dec_b1 + (size_t)i * NB_F,
             Hf, M, NB_D, NB_F, 1, stream);
        gemm(Hf, dec_w2 + (size_t)i * NB_F * NB_D, dec_b2 + (size_t)i * NB_D,
             T1, M, NB_F, NB_D, 0, stream);
        float* outp = (i == NB_L - 1) ? (float*)d_out : Y;
        add_ln_kernel<<<dim3(M), 256, 0, stream>>>(Y, T1, ls + 2 * NB_D,
                                                   lb + 2 * NB_D, outp, 1.0f);
    }
}
